// LlamaAttention_14078902796655
// MI455X (gfx1250) — matrix operand order repaired and checked by reading, not compiled, not run on hardware
//
#include <hip/hip_runtime.h>
#include <hip/hip_bf16.h>
#include <stddef.h>

// ---------------- problem constants ----------------
#define BATCH     2
#define SEQ       2048
#define HIDDEN    2048
#define NHEADS    32
#define NKV       8
#define HDIM      64
#define QKVN      3072              // 2048 (Q) + 512 (K) + 512 (V)
#define ROWS      (BATCH*SEQ)       // 4096

#define DEVI __device__ __forceinline__

typedef __attribute__((ext_vector_type(16))) __bf16 v16bf;
typedef __attribute__((ext_vector_type(8)))  float  v8f;

union Frag16 { v16bf v; unsigned u[8]; };

DEVI v8f wmma_bf16(v16bf a, v16bf b, v8f c) {
  // (neg_a, A, neg_b, B, c_mod, C, reuse_a, reuse_b)
  return __builtin_amdgcn_wmma_f32_16x16x32_bf16(false, a, false, b, (short)0, c, false, false);
}

DEVI v8f zero8() { v8f z = {0.f,0.f,0.f,0.f,0.f,0.f,0.f,0.f}; return z; }

DEVI __bf16 bf_lo(unsigned u) { return __builtin_bit_cast(__bf16, (unsigned short)(u & 0xffffu)); }
DEVI __bf16 bf_hi(unsigned u) { return __builtin_bit_cast(__bf16, (unsigned short)(u >> 16)); }

// Store the 8 bf16 halves of a uint4 to dst[0], dst[ld], ..., dst[7*ld]
// (transpose scatter; backend uses ds_store_b16 / ds_store_b16_d16_hi from regs).
DEVI void trstore8(__bf16* dst, int ld, uint4 x) {
  dst[0 * ld] = bf_lo(x.x); dst[1 * ld] = bf_hi(x.x);
  dst[2 * ld] = bf_lo(x.y); dst[3 * ld] = bf_hi(x.y);
  dst[4 * ld] = bf_lo(x.z); dst[5 * ld] = bf_hi(x.z);
  dst[6 * ld] = bf_lo(x.w); dst[7 * ld] = bf_hi(x.w);
}

// A-fragment (16x32, MxK) from a row-major bf16 matrix with leading dim ld.
// Contiguous k order used here (same as load_bT): lane L -> row = L%16; lanes<16 hold K {0..15}, lanes>=16 hold {16..31}.
// The 8 dword reads per lane are two contiguous 16B runs -> compiler merges to ds_load_b128.
DEVI v16bf load_a_rm(const __bf16* p, int ld, int lane) {
  Frag16 f;
  const int row = lane & 15;
  const int kh  = (lane >> 4) << 3;              // 0 or 8
  const __bf16* rp = p + (size_t)row * ld;
#pragma unroll
  for (int v = 0; v < 8; ++v) {
    const int k = 2 * v + 2 * kh;
    f.u[v] = *(const unsigned*)(rp + k);
  }
  return f.v;
}

// B-fragment (32x16, KxN) taken from B^T stored row-major (bt[n][k], leading dim ld).
// Contiguous k order used here (same as load_a_rm): lane L -> col n = L%16; elements e -> K = k0 + 16*(L/16) + e (sequential),
// contiguous along a row of B^T -> two uint4 loads per lane.
DEVI v16bf load_bT(const __bf16* bt, int ld, int lane, int n0, int k0) {
  const __bf16* p = bt + (size_t)(n0 + (lane & 15)) * ld + k0 + ((lane >> 4) << 4);
  Frag16 f;
  const uint4* q = (const uint4*)p;
  uint4 x0 = q[0], x1 = q[1];
  f.u[0] = x0.x; f.u[1] = x0.y; f.u[2] = x0.z; f.u[3] = x0.w;
  f.u[4] = x1.x; f.u[5] = x1.y; f.u[6] = x1.z; f.u[7] = x1.w;
  return f.v;
}

// ---------------- conversion / packing kernels ----------------
__global__ void conv_f32_bf16_kernel(const float* __restrict__ src,
                                     __bf16* __restrict__ dst, long n) {
  long i = (long)blockIdx.x * blockDim.x + threadIdx.x;
  if (i < n) dst[i] = (__bf16)src[i];
}

__global__ void pack_qkv_w_kernel(const float* __restrict__ Wq,
                                  const float* __restrict__ Wk,
                                  const float* __restrict__ Wv,
                                  __bf16* __restrict__ out) {
  long i = (long)blockIdx.x * blockDim.x + threadIdx.x;
  const long total = (long)HIDDEN * QKVN;
  if (i >= total) return;
  int row = (int)(i / QKVN);
  int col = (int)(i % QKVN);
  float v;
  if (col < 2048)       v = Wq[(size_t)row * 2048 + col];
  else if (col < 2560)  v = Wk[(size_t)row * 512 + (col - 2048)];
  else                  v = Wv[(size_t)row * 512 + (col - 2560)];
  out[i] = (__bf16)v;
}

// ---------------- WMMA GEMM: C[M,N] = A[M,K] @ B[K,N], bf16 in, f32 acc ----------------
// block = 128 threads (4 waves), block tile 128x64, wave tile 32x64, K step 32.
// Software-pipelined with NAMED uint4 registers (no arrays / no address-of -> no spills).
__global__ __launch_bounds__(128) void gemm_bf16_kernel(
    const __bf16* __restrict__ A, const __bf16* __restrict__ Bm,
    float* __restrict__ Cf, __bf16* __restrict__ Cb, int M, int N, int K) {
  const int n0   = blockIdx.x * 64;
  const int m0b  = blockIdx.y * 128;
  const int tid  = threadIdx.x;
  const int lane = tid & 31;
  const int wave = tid >> 5;

  __shared__ __bf16 As[128 * 32];    // A tile, row-major, ld=32
  __shared__ __bf16 Bt[64 * 40];     // B tile transposed: Bt[n][k], ld=40 (pad)

  v8f acc[2][4];
#pragma unroll
  for (int s = 0; s < 2; ++s)
#pragma unroll
    for (int j = 0; j < 4; ++j) acc[s][j] = zero8();

  const int arow = tid;                              // A: one 32-half row per thread
  const int brow = tid >> 2, boff = (tid & 3) * 16;  // B: 16 halves per thread

  const __bf16* aptr = A + (size_t)(m0b + arow) * K;
  const __bf16* bptr = Bm + (size_t)brow * N + n0 + boff;

  uint4 a0r, a1r, a2r, a3r, b0r, b1r;
  {
    const uint4* ga = (const uint4*)aptr;
    a0r = ga[0]; a1r = ga[1]; a2r = ga[2]; a3r = ga[3];
    const uint4* gb = (const uint4*)bptr;
    b0r = gb[0]; b1r = gb[1];
  }

  __bf16* asd = As + arow * 32;
  __bf16* bsd = Bt + boff * 40 + brow;

  for (int k0 = 0; k0 < K; k0 += 32) {
    __syncthreads();                                  // previous compute done
    {
      uint4* da = (uint4*)asd;
      da[0] = a0r; da[1] = a1r; da[2] = a2r; da[3] = a3r;
      trstore8(bsd,            40, b0r);
      trstore8(bsd + 8 * 40,   40, b1r);
    }
    __syncthreads();                                  // tiles visible
    if (k0 + 32 < K) {                                // overlap VMEM with WMMA
      const uint4* ga = (const uint4*)(aptr + k0 + 32);
      a0r = ga[0]; a1r = ga[1]; a2r = ga[2]; a3r = ga[3];
      const uint4* gb = (const uint4*)(bptr + (size_t)(k0 + 32) * N);
      b0r = gb[0]; b1r = gb[1];
    }

#pragma unroll
    for (int s = 0; s < 2; ++s) {
      v16bf a = load_a_rm(As + (wave * 32 + s * 16) * 32, 32, lane);
      acc[s][0] = wmma_bf16(a, load_bT(Bt, 40, lane, 0,  0), acc[s][0]);
      acc[s][1] = wmma_bf16(a, load_bT(Bt, 40, lane, 16, 0), acc[s][1]);
      acc[s][2] = wmma_bf16(a, load_bT(Bt, 40, lane, 32, 0), acc[s][2]);
      acc[s][3] = wmma_bf16(a, load_bT(Bt, 40, lane, 48, 0), acc[s][3]);
    }
  }

  const int ncol = lane & 15;
  const int hsel = lane >> 4;
#pragma unroll
  for (int s = 0; s < 2; ++s) {
#pragma unroll
    for (int r = 0; r < 8; ++r) {
      const int row = m0b + wave * 32 + s * 16 + r + 8 * hsel;
      const size_t base = (size_t)row * N + n0;
      if (Cf) {
        Cf[base + ncol]      = acc[s][0][r];
        Cf[base + 16 + ncol] = acc[s][1][r];
        Cf[base + 32 + ncol] = acc[s][2][r];
        Cf[base + 48 + ncol] = acc[s][3][r];
      } else {
        Cb[base + ncol]      = (__bf16)acc[s][0][r];
        Cb[base + 16 + ncol] = (__bf16)acc[s][1][r];
        Cb[base + 32 + ncol] = (__bf16)acc[s][2][r];
        Cb[base + 48 + ncol] = (__bf16)acc[s][3][r];
      }
    }
  }
}

// ---------------- RoPE + head-major scatter ----------------
__global__ void rope_kernel(const __bf16* __restrict__ QKV,
                            const float* __restrict__ cosb,
                            const float* __restrict__ sinb,
                            __bf16* __restrict__ Qo, __bf16* __restrict__ Ko,
                            __bf16* __restrict__ Vo) {
  long i = (long)blockIdx.x * blockDim.x + threadIdx.x;
  const long total = (long)ROWS * QKVN;
  if (i >= total) return;
  const int row = (int)(i / QKVN);
  const int c   = (int)(i % QKVN);
  const int b   = row >> 11;            // /2048
  const int s   = row & 2047;
  const int d   = c & 63;
  float val = (float)QKV[i];
  if (c < 2560) {
    const int pc = (d < 32) ? (c + 32) : (c - 32);
    float pv  = (float)QKV[(size_t)row * QKVN + pc];
    float rot = (d < 32) ? -pv : pv;
    const size_t cidx = (size_t)(b * SEQ + s) * HDIM + d;
    val = val * cosb[cidx] + rot * sinb[cidx];
    if (c < 2048) {
      const int h = c >> 6;
      Qo[(((size_t)b * NHEADS + h) * SEQ + s) * HDIM + d] = (__bf16)val;
    } else {
      const int kh = (c - 2048) >> 6;
      Ko[(((size_t)b * NKV + kh) * SEQ + s) * HDIM + d] = (__bf16)val;
    }
  } else {
    const int vh = (c - 2560) >> 6;
    Vo[(((size_t)b * NKV + vh) * SEQ + s) * HDIM + d] = (__bf16)val;
  }
}

// ---------------- causal flash attention ----------------
// grid = (S/64, NHEADS, BATCH), block = 128 (4 waves, one 16-row Q tile per wave).
// 64-key chunks: 8 score WMMAs + 8 PV WMMAs per barrier pair.
__global__ __launch_bounds__(128) void attn_kernel(
    const __bf16* __restrict__ Qb, const __bf16* __restrict__ Kb,
    const __bf16* __restrict__ Vb, __bf16* __restrict__ Ob) {
  const int tile = blockIdx.x;
  const int h    = blockIdx.y;
  const int b    = blockIdx.z;
  const int kvh  = h >> 2;                       // 32/8 = 4-way GQA
  const int tid  = threadIdx.x;
  const int lane = tid & 31;
  const int wave = tid >> 5;
  const int m0   = tile * 64 + wave * 16;

  const __bf16* Qh = Qb + ((size_t)b * NHEADS + h) * SEQ * HDIM;
  const __bf16* Kh = Kb + ((size_t)b * NKV + kvh)  * SEQ * HDIM;
  const __bf16* Vh = Vb + ((size_t)b * NKV + kvh)  * SEQ * HDIM;

  __shared__ __bf16 Ks[64 * 64];                 // key chunk, row-major (= B^T for scores)
  __shared__ __bf16 Vt[64 * 72];                 // value chunk transposed: Vt[d][k], pad 72
  __shared__ __bf16 Ps[4][16 * 64];              // per-wave P tile (16x64) for transpose

  // Q fragments held in registers for the whole pass (16x64 = two 16x32 A frags).
  v16bf qa0 = load_a_rm(Qh + (size_t)m0 * HDIM,      HDIM, lane);
  v16bf qa1 = load_a_rm(Qh + (size_t)m0 * HDIM + 32, HDIM, lane);

  v8f o[4];
#pragma unroll
  for (int j = 0; j < 4; ++j) o[j] = zero8();
  float mst[8], lst[8];
#pragma unroll
  for (int r = 0; r < 8; ++r) { mst[r] = -3.0e38f; lst[r] = 0.f; }

  const int ncol = lane & 15;
  const int hsel = lane >> 4;
  const int lrow = tid >> 1, loff = (tid & 1) * 32;  // loader: 32 halves/thread

  const __bf16* kptr = Kh + (size_t)lrow * HDIM + loff;
  const __bf16* vptr = Vh + (size_t)lrow * HDIM + loff;
  __bf16* ksd = Ks + lrow * 64 + loff;
  __bf16* vsd = Vt + loff * 72 + lrow;

  const int nchunks = tile + 1;                  // causal: keys < (tile+1)*64
  for (int c = 0; c < nchunks; ++c) {
    const int kbase = c * 64;
    __syncthreads();
    {
      const size_t goff = (size_t)kbase * HDIM;
      const uint4* gk = (const uint4*)(kptr + goff);
      uint4 k0 = gk[0], k1 = gk[1], k2 = gk[2], k3 = gk[3];
      uint4* dk = (uint4*)ksd;
      dk[0] = k0; dk[1] = k1; dk[2] = k2; dk[3] = k3;
      const uint4* gv = (const uint4*)(vptr + goff);
      uint4 v0 = gv[0], v1 = gv[1], v2 = gv[2], v3 = gv[3];
      trstore8(vsd,           72, v0);
      trstore8(vsd +  8 * 72, 72, v1);
      trstore8(vsd + 16 * 72, 72, v2);
      trstore8(vsd + 24 * 72, 72, v3);
      if (c + 1 < nchunks) {
        __builtin_prefetch(kptr + goff + 64 * HDIM, 0, 1);
        __builtin_prefetch(vptr + goff + 64 * HDIM, 0, 1);
      }
    }
    __syncthreads();

    // scores: four 16x16 tiles over this 64-key chunk, K-dim = 64 (two wmma each)
    v8f s[4];
#pragma unroll
    for (int t = 0; t < 4; ++t) {
      s[t] = wmma_bf16(qa0, load_bT(Ks, 64, lane, 16 * t, 0),  zero8());
      s[t] = wmma_bf16(qa1, load_bT(Ks, 64, lane, 16 * t, 32), s[t]);
    }

    const bool diag = (c == tile);               // only last chunk crosses the diagonal
    __bf16* P = Ps[wave];
#pragma unroll
    for (int r = 0; r < 8; ++r) {
      const int mrow = m0 + r + 8 * hsel;
      float v0 = s[0][r] * 0.125f;
      float v1 = s[1][r] * 0.125f;
      float v2 = s[2][r] * 0.125f;
      float v3 = s[3][r] * 0.125f;
      if (diag) {
        if (kbase +      ncol > mrow) v0 += -1.0e9f;
        if (kbase + 16 + ncol > mrow) v1 += -1.0e9f;
        if (kbase + 32 + ncol > mrow) v2 += -1.0e9f;
        if (kbase + 48 + ncol > mrow) v3 += -1.0e9f;
      }
      float mx = fmaxf(fmaxf(v0, v1), fmaxf(v2, v3));
#pragma unroll
      for (int sh = 1; sh < 16; sh <<= 1) mx = fmaxf(mx, __shfl_xor(mx, sh, 32));
      const float mnew  = fmaxf(mst[r], mx);
      const float alpha = __expf(mst[r] - mnew);
      const float p0 = __expf(v0 - mnew);
      const float p1 = __expf(v1 - mnew);
      const float p2 = __expf(v2 - mnew);
      const float p3 = __expf(v3 - mnew);
      const int prow = (r + 8 * hsel) * 64;
      P[prow + ncol]      = (__bf16)p0;          // C-layout -> LDS for A-frag transpose
      P[prow + 16 + ncol] = (__bf16)p1;
      P[prow + 32 + ncol] = (__bf16)p2;
      P[prow + 48 + ncol] = (__bf16)p3;
      float sum = (p0 + p1) + (p2 + p3);
#pragma unroll
      for (int sh = 1; sh < 16; sh <<= 1) sum += __shfl_xor(sum, sh, 32);
      lst[r] = lst[r] * alpha + sum;
      mst[r] = mnew;
#pragma unroll
      for (int j = 0; j < 4; ++j) o[j][r] *= alpha;
    }

    // P (16x64) as two A-fragments; V chunk (64x64) as 4x2 B-fragments
    v16bf pa0 = load_a_rm(P,      64, lane);
    v16bf pa1 = load_a_rm(P + 32, 64, lane);
#pragma unroll
    for (int j = 0; j < 4; ++j) {
      o[j] = wmma_bf16(pa0, load_bT(Vt, 72, lane, 16 * j, 0),  o[j]);
      o[j] = wmma_bf16(pa1, load_bT(Vt, 72, lane, 16 * j, 32), o[j]);
    }
  }

  // normalize and scatter to [b, s, h*64+d] bf16 (input of the Wo GEMM)
#pragma unroll
  for (int r = 0; r < 8; ++r) {
    const float inv = 1.0f / lst[r];
    const int srow = m0 + r + 8 * hsel;
    __bf16* dst = Ob + ((size_t)(b * SEQ + srow)) * (NHEADS * HDIM) + h * HDIM;
    dst[ncol]      = (__bf16)(o[0][r] * inv);
    dst[16 + ncol] = (__bf16)(o[1][r] * inv);
    dst[32 + ncol] = (__bf16)(o[2][r] * inv);
    dst[48 + ncol] = (__bf16)(o[3][r] * inv);
  }
}

// ---------------- host launcher ----------------
extern "C" void kernel_launch(void* const* d_in, const int* in_sizes, int n_in,
                              void* d_out, int out_size, void* d_ws, size_t ws_size,
                              hipStream_t stream) {
  const float* hidden = (const float*)d_in[0];
  // d_in[1] = attention_mask (pure causal; computed analytically, not read)
  const float* cosb   = (const float*)d_in[2];
  const float* sinb   = (const float*)d_in[3];
  const float* Wq     = (const float*)d_in[4];
  const float* Wk     = (const float*)d_in[5];
  const float* Wv     = (const float*)d_in[6];
  const float* Wo     = (const float*)d_in[7];
  float* out = (float*)d_out;

  char* ws = (char*)d_ws;
  size_t off = 0;
  auto carve = [&](size_t bytes) { char* p = ws + off; off += (bytes + 255) & ~(size_t)255; return p; };

  __bf16* Xb    = (__bf16*)carve((size_t)ROWS * HIDDEN * 2);
  __bf16* Wqkvb = (__bf16*)carve((size_t)HIDDEN * QKVN * 2);
  __bf16* Wob   = (__bf16*)carve((size_t)HIDDEN * HIDDEN * 2);
  __bf16* QKVb  = (__bf16*)carve((size_t)ROWS * QKVN * 2);
  __bf16* Qh    = (__bf16*)carve((size_t)BATCH * NHEADS * SEQ * HDIM * 2);
  __bf16* Kh    = (__bf16*)carve((size_t)BATCH * NKV * SEQ * HDIM * 2);
  __bf16* Vh    = (__bf16*)carve((size_t)BATCH * NKV * SEQ * HDIM * 2);
  __bf16* Attnb = (__bf16*)carve((size_t)ROWS * HIDDEN * 2);
  (void)ws_size;

  const int T = 256;
  {
    long n = (long)ROWS * HIDDEN;
    conv_f32_bf16_kernel<<<dim3((unsigned)((n + T - 1) / T)), T, 0, stream>>>(hidden, Xb, n);
  }
  {
    long n = (long)HIDDEN * QKVN;
    pack_qkv_w_kernel<<<dim3((unsigned)((n + T - 1) / T)), T, 0, stream>>>(Wq, Wk, Wv, Wqkvb);
  }
  {
    long n = (long)HIDDEN * HIDDEN;
    conv_f32_bf16_kernel<<<dim3((unsigned)((n + T - 1) / T)), T, 0, stream>>>(Wo, Wob, n);
  }
  // fused QKV projection (bf16 out)
  gemm_bf16_kernel<<<dim3(QKVN / 64, ROWS / 128), 128, 0, stream>>>(
      Xb, Wqkvb, nullptr, QKVb, ROWS, QKVN, HIDDEN);
  // RoPE + scatter to head-major
  {
    long n = (long)ROWS * QKVN;
    rope_kernel<<<dim3((unsigned)((n + T - 1) / T)), T, 0, stream>>>(QKVb, cosb, sinb, Qh, Kh, Vh);
  }
  // causal flash attention
  attn_kernel<<<dim3(SEQ / 64, NHEADS, BATCH), 128, 0, stream>>>(Qh, Kh, Vh, Attnb);
  // output projection (f32 out)
  gemm_bf16_kernel<<<dim3(HIDDEN / 64, ROWS / 128), 128, 0, stream>>>(
      Attnb, Wob, out, nullptr, ROWS, HIDDEN, HIDDEN);
  (void)out_size; (void)n_in; (void)in_sizes;
}
